// EncoderBlock_8160437862870
// MI455X (gfx1250) — matrix operand order repaired and checked by reading, not compiled, not run on hardware
//
#include <hip/hip_runtime.h>
#include <math.h>
#include <stdint.h>

// ---------------------------------------------------------------------------
// Problem constants (from the reference)
// ---------------------------------------------------------------------------
constexpr int kB   = 2;
constexpr int kS   = 2048;
constexpr int kD   = 1024;
constexpr int kH   = 16;
constexpr int kDK  = 64;
constexpr int kDFF = 4096;
constexpr int kM   = kB * kS;      // 4096 token rows
constexpr float kEps = 1e-5f;

typedef __attribute__((ext_vector_type(16))) _Float16 v16h;
typedef __attribute__((ext_vector_type(8)))  _Float16 v8h;
typedef __attribute__((ext_vector_type(4)))  _Float16 v4h;
typedef __attribute__((ext_vector_type(8)))  float    v8f;
typedef __attribute__((ext_vector_type(4)))  float    v4f;
typedef __attribute__((ext_vector_type(4)))  unsigned int u32x4;
typedef __attribute__((ext_vector_type(4)))  int          i32x4;
typedef __attribute__((ext_vector_type(8)))  int          i32x8;

#if defined(__has_builtin)
#if __has_builtin(__builtin_amdgcn_tensor_load_to_lds)
#define ENC_HAS_TDM 1
#endif
#endif

__device__ __forceinline__ v8f zero8() {
  v8f z;
#pragma unroll
  for (int i = 0; i < 8; ++i) z[i] = 0.0f;
  return z;
}

// ---------------------------------------------------------------------------
// XOR-lane reduction step via DS_SWIZZLE_B32 (group-of-32 mode):
// offset = {0, xor_mask[14:10], or_mask[9:5]=0, and_mask[4:0]=0x1F}
// ---------------------------------------------------------------------------
template <int MASK>
__device__ __forceinline__ float swz_xor(float v) {
  return __int_as_float(
      __builtin_amdgcn_ds_swizzle(__float_as_int(v), (MASK << 10) | 0x1F));
}

// ---------------------------------------------------------------------------
// CDNA5 async global->LDS staging (CDNA5 ISA).
// ---------------------------------------------------------------------------
__device__ __forceinline__ uint32_t lds_off(const void* p) {
  // generic (flat) shared pointers carry the LDS offset in the low 32 bits
  return (uint32_t)(uintptr_t)p;
}

__device__ __forceinline__ void async_copy_b128(uint32_t dst_lds, const void* src) {
  asm volatile("global_load_async_to_lds_b128 %0, %1, off"
               :: "v"(dst_lds), "v"(src) : "memory");
}

template <int N>
__device__ __forceinline__ void wait_asynccnt() {
  asm volatile("s_wait_asynccnt %0" :: "i"(N) : "memory");
}

// ---------------------------------------------------------------------------
// Tensor Data Mover: 2D f16 tile load, global -> LDS, with LDS padding that
// reproduces a (tile_k + 8)-half row pitch (pad 4 DWORDs every 16 DWORDs).
// Descriptor layout per CDNA5 ISA (D# groups 0/1).
// ---------------------------------------------------------------------------
#ifdef ENC_HAS_TDM
__device__ __forceinline__ void tdm_load_2d_f16(uint32_t lds_byte_addr,
                                                const void* gptr,
                                                uint32_t row_stride_elems,
                                                uint32_t tile_k,
                                                uint32_t tile_rows) {
  const uint64_t ga = (uint64_t)(uintptr_t)gptr;
  u32x4 g0;
  g0[0] = 1u;                                            // count=1, user desc
  g0[1] = lds_byte_addr;                                 // lds_addr [63:32]
  g0[2] = (uint32_t)ga;                                  // global_addr lo
  g0[3] = (uint32_t)((ga >> 32) & 0x1FFFFFFu) | (2u << 30); // ga hi | type=2
  i32x8 g1;
  g1[0] = (int)((1u << 16)      // data_size = 1 -> 2-byte elements
              | (1u << 20)      // pad_enable
              | (3u << 22)      // pad_interval: 16 DWORDs between pads
              | (3u << 25));    // pad_amount:   4 DWORDs (8 halves)
  const uint32_t td0 = 0x7FFFFFFFu;   // huge tensor dims: no OOB clipping
  const uint32_t td1 = 0x7FFFFFFFu;
  g1[1] = (int)((td0 & 0xFFFFu) << 16);                  // tensor_dim0 lo16
  g1[2] = (int)((td0 >> 16) | ((td1 & 0xFFFFu) << 16));  // dim0 hi | dim1 lo
  g1[3] = (int)((td1 >> 16) | (tile_k << 16));           // dim1 hi | tile_dim0
  g1[4] = (int)(tile_rows);                              // tile_dim1 | tile_dim2=0
  g1[5] = (int)row_stride_elems;                         // tensor_dim0_stride lo32
  g1[6] = 0;                                             // stride0 hi | stride1 lo
  g1[7] = 0;
  const i32x4 z4 = {0, 0, 0, 0};
#if __clang_major__ >= 23
  const i32x8 z8 = {0, 0, 0, 0, 0, 0, 0, 0};
  __builtin_amdgcn_tensor_load_to_lds(g0, g1, z4, z4, z8, 0);
#else
  __builtin_amdgcn_tensor_load_to_lds(g0, g1, z4, z4, 0);
#endif
}
#endif

// ---------------------------------------------------------------------------
// WMMA fragment loaders (layouts per CDNA5 ISA §7.12.2, wave32).
// Every fragment is exactly two (ds_)load_b128's.
// ---------------------------------------------------------------------------
__device__ __forceinline__ v16h frag_from(const _Float16* p0, const _Float16* p1) {
  v8h lo = *(const v8h*)p0;
  v8h hi = *(const v8h*)p1;
  return __builtin_shufflevector(lo, hi, 0, 1, 2, 3, 4, 5, 6, 7,
                                 8, 9, 10, 11, 12, 13, 14, 15);
}

// A fragment: 16(M) x 32(K) f16 from a row-major tile, leading dim ld.
__device__ __forceinline__ v16h load_a_frag(const _Float16* __restrict__ src, int ld) {
  const int lane = threadIdx.x & 31;
  const int m    = lane & 15;
  const int koff = (lane >> 4) << 3;
  const _Float16* r = src + (size_t)m * ld + koff;
  return frag_from(r, r + 16);
}

// B fragment: logical B[k][n] read from a tile stored as [n rows][k cols]
// row-major with leading dim ld (N-major / K-contiguous layout).
__device__ __forceinline__ v16h load_b_fragNK(const _Float16* __restrict__ src, int ld) {
  const int lane = threadIdx.x & 31;
  const int n    = lane & 15;
  const int koff = (lane >> 4) << 3;
  const _Float16* r = src + (size_t)n * ld + koff;
  return frag_from(r, r + 16);
}

__device__ __forceinline__ v8f wmma_f16(v16h a, v16h b, v8f c) {
  return __builtin_amdgcn_wmma_f32_16x16x32_f16(false, a, false, b, (short)0, c,
                                                false, false);
}

// ---------------------------------------------------------------------------
// fp32 -> fp16 conversion (activations)
// ---------------------------------------------------------------------------
__global__ void enc_cvt_f16_kernel(const float* __restrict__ src,
                                   _Float16* __restrict__ dst, int n) {
  int i = blockIdx.x * blockDim.x + threadIdx.x;
  if (i < n) dst[i] = (_Float16)src[i];
}

// ---------------------------------------------------------------------------
// fp32 W[K][N]  ->  f16 Wt[N][K]   (tiled 32x32 transpose through LDS)
// ---------------------------------------------------------------------------
__global__ void __launch_bounds__(256)
enc_transpose_cvt_kernel(const float* __restrict__ W, _Float16* __restrict__ Wt,
                         int K, int N) {
  __shared__ _Float16 T[32][33];
  const int c  = threadIdx.x & 31;
  const int r0 = threadIdx.x >> 5;           // 0..7
  const int k0 = blockIdx.y * 32;
  const int n0 = blockIdx.x * 32;
#pragma unroll
  for (int i = 0; i < 4; ++i) {
    const int r = r0 + i * 8;
    T[r][c] = (_Float16)W[(size_t)(k0 + r) * N + n0 + c];
  }
  __syncthreads();
#pragma unroll
  for (int i = 0; i < 4; ++i) {
    const int r = r0 + i * 8;
    Wt[(size_t)(n0 + r) * K + k0 + c] = T[c][r];
  }
}

// ---------------------------------------------------------------------------
// Tiled WMMA GEMM:  C[M,N] = A[M,K](f16 row) * W[K,N]  with W as Wt[N][K].
// Block tile 128x128, 8 waves (2x4), each wave 64x32 (4x2 WMMA tiles).
// K chunks of 32, double-buffered LDS.  Staging: TDM (wave 0 issues two 2D
// tensor descriptors per buffer; hardware applies the 8-half row padding),
// falling back to async b128 copies if the TDM builtin is unavailable.
// OUT_MODE: 0 = f32 row-major, 1 = f16 row-major, 2 = f16 V-transposed.
// SCALE8 : multiply the result by 0.125 (pre-scales Q by 1/sqrt(DK)).
// ---------------------------------------------------------------------------
template <int HAS_BIAS, int RELU, int OUT_MODE, int SCALE8>
__global__ void __launch_bounds__(256)
enc_gemm_kernel(const _Float16* __restrict__ A, const _Float16* __restrict__ Wt,
                const float* __restrict__ bias, void* __restrict__ Cout,
                int M, int N, int K) {
  constexpr int LDA = 40;   // 32 + 8 pad (halves) == TDM pad output pitch
  constexpr int LDB = 40;
  __shared__ _Float16 As[2][128 * LDA];
  __shared__ _Float16 Bs[2][128 * LDB];

  const int tid  = threadIdx.x;
  const int wave = tid >> 5;
  const int lane = tid & 31;
  const int wrow = wave >> 2;                  // 0..1  (M direction)
  const int wcol = wave & 3;                   // 0..3  (N direction)
  const int m0   = blockIdx.y * 128;
  const int n0   = blockIdx.x * 128;

#ifdef ENC_HAS_TDM
  const bool stager = (wave == 0);             // wave-uniform predicate
  auto stage = [&](int buf, int k0) {
    if (stager) {
      tdm_load_2d_f16(lds_off(&As[buf][0]), A + (size_t)m0 * K + k0, K, 32, 128);
      tdm_load_2d_f16(lds_off(&Bs[buf][0]), Wt + (size_t)n0 * K + k0, K, 32, 128);
    }
  };
#else
  auto stage = [&](int buf, int k0) {
#pragma unroll
    for (int idx = tid; idx < 128 * 4; idx += 256) {
      const int r = idx >> 2, cg = (idx & 3) * 8;
      async_copy_b128(lds_off(&As[buf][r * LDA + cg]),
                      A + (size_t)(m0 + r) * K + k0 + cg);
      async_copy_b128(lds_off(&Bs[buf][r * LDB + cg]),
                      Wt + (size_t)(n0 + r) * K + k0 + cg);
    }
  };
#endif

  v8f acc[4][2];
#pragma unroll
  for (int i = 0; i < 4; ++i)
#pragma unroll
    for (int j = 0; j < 2; ++j) acc[i][j] = zero8();

  stage(0, 0);
  int cur = 0;
  for (int k0 = 0; k0 < K; k0 += 32) {
    const bool more = (k0 + 32) < K;
#ifdef ENC_HAS_TDM
    if (more) stage(cur ^ 1, k0 + 32);
    if (stager) {
      if (more) __builtin_amdgcn_s_wait_tensorcnt(2);  // prev pair complete
      else      __builtin_amdgcn_s_wait_tensorcnt(0);
    }
#else
    if (more) { stage(cur ^ 1, k0 + 32); wait_asynccnt<4>(); }
    else      { wait_asynccnt<0>(); }
#endif
    __syncthreads();

    const _Float16* as = As[cur];
    const _Float16* bs = Bs[cur];
    v16h bf[2], af[4];
#pragma unroll
    for (int j = 0; j < 2; ++j)
      bf[j] = load_b_fragNK(bs + (wcol * 32 + j * 16) * LDB, LDB);
#pragma unroll
    for (int i = 0; i < 4; ++i)
      af[i] = load_a_frag(as + (wrow * 64 + i * 16) * LDA, LDA);
#pragma unroll
    for (int i = 0; i < 4; ++i)
#pragma unroll
      for (int j = 0; j < 2; ++j) acc[i][j] = wmma_f16(af[i], bf[j], acc[i][j]);
    __syncthreads();
    cur ^= 1;
  }

  // epilogue: C layout -> lane holds col (lane&15), rows r + 8*(lane>=16)
  const int nn    = lane & 15;
  const int lhalf = lane >> 4;
#pragma unroll
  for (int i = 0; i < 4; ++i) {
#pragma unroll
    for (int j = 0; j < 2; ++j) {
      const int col = n0 + wcol * 32 + j * 16 + nn;
      const float bv = HAS_BIAS ? bias[col] : 0.0f;
#pragma unroll
      for (int r = 0; r < 8; ++r) {
        const int row = m0 + wrow * 64 + i * 16 + r + lhalf * 8;
        float v = acc[i][j][r] + bv;
        if (RELU) v = fmaxf(v, 0.0f);
        if (SCALE8) v *= 0.125f;
        if (OUT_MODE == 0) {
          ((float*)Cout)[(size_t)row * N + col] = v;
        } else if (OUT_MODE == 1) {
          ((_Float16*)Cout)[(size_t)row * N + col] = (_Float16)v;
        } else {
          const int bb = row / kS, s = row % kS;
          const int hh = col / kDK, dd = col % kDK;
          ((_Float16*)Cout)[(((size_t)bb * kH + hh) * kDK + dd) * kS + s] =
              (_Float16)v;
        }
      }
    }
  }
}

// ---------------------------------------------------------------------------
// Flash attention: grid = (S/128, H, B), 8 waves; wave owns 16 query rows.
// 64-key chunks, double-buffered async staging, Q pre-scaled by 1/sqrt(DK),
// online softmax with ds_swizzle xor reductions, 16 WMMAs per chunk.
// ---------------------------------------------------------------------------
__global__ void __launch_bounds__(256)
enc_attention_kernel(const _Float16* __restrict__ Q, const _Float16* __restrict__ Kg,
                     const _Float16* __restrict__ Vt, _Float16* __restrict__ O) {
  constexpr int LDK = 72;   // 64 + 8 pad (halves); rows stay 16B aligned
  constexpr int LDV = 72;
  constexpr int LDP = 72;
  __shared__ _Float16 Ks[2][64 * LDK];
  __shared__ _Float16 Vs[2][64 * LDV];
  __shared__ _Float16 Ps[8][16 * LDP];

  const int tid  = threadIdx.x;
  const int wave = tid >> 5;
  const int lane = tid & 31;
  const int b  = blockIdx.z;
  const int h  = blockIdx.y;
  const int q0 = blockIdx.x * 128 + wave * 16;

  const _Float16* qbase = Q + ((size_t)(b * kS) + q0) * kD + h * kDK;
  const v16h aq0 = load_a_frag(qbase, kD);        // d 0..31 (pre-scaled)
  const v16h aq1 = load_a_frag(qbase + 32, kD);   // d 32..63

  const size_t vrow = ((size_t)b * kH + h) * kDK; // first d-row of head in Vt

  auto stage = [&](int buf, int k0) {
#pragma unroll
    for (int idx = tid; idx < 64 * 8; idx += 256) {   // 2 iterations
      const int r = idx >> 3, cg = (idx & 7) * 8;
      async_copy_b128(lds_off(&Ks[buf][r * LDK + cg]),
                      Kg + ((size_t)(b * kS) + k0 + r) * kD + h * kDK + cg);
      async_copy_b128(lds_off(&Vs[buf][r * LDV + cg]),
                      Vt + (vrow + r) * kS + k0 + cg);
    }
  };

  v8f acc[4];
#pragma unroll
  for (int j = 0; j < 4; ++j) acc[j] = zero8();
  float mrow[8], lrow[8];
#pragma unroll
  for (int r = 0; r < 8; ++r) { mrow[r] = -INFINITY; lrow[r] = 0.0f; }

  _Float16* pmine = &Ps[wave][0];
  const int nn = lane & 15;

  stage(0, 0);
  int cur = 0;
  for (int k0 = 0; k0 < kS; k0 += 64) {
    const bool more = (k0 + 64) < kS;
    if (more) { stage(cur ^ 1, k0 + 64); wait_asynccnt<4>(); }
    else      { wait_asynccnt<0>(); }
    __syncthreads();

    // scores: four 16x16 tiles over keys [k0 .. k0+64)
    v8f sc[4];
#pragma unroll
    for (int t = 0; t < 4; ++t) {
      v16h b0 = load_b_fragNK(&Ks[cur][(t * 16) * LDK], LDK);        // d 0..31
      v16h b1 = load_b_fragNK(&Ks[cur][(t * 16) * LDK + 32], LDK);   // d 32..63
      v8f c = zero8();
      c = wmma_f16(aq0, b0, c);
      c = wmma_f16(aq1, b1, c);
      sc[t] = c;
    }

    // online softmax per row (row values live across 16 lanes of each half)
#pragma unroll
    for (int r = 0; r < 8; ++r) {
      const float s0 = sc[0][r], s1 = sc[1][r];
      const float s2 = sc[2][r], s3 = sc[3][r];
      float vmax = fmaxf(fmaxf(s0, s1), fmaxf(s2, s3));
      vmax = fmaxf(vmax, swz_xor<1>(vmax));
      vmax = fmaxf(vmax, swz_xor<2>(vmax));
      vmax = fmaxf(vmax, swz_xor<4>(vmax));
      vmax = fmaxf(vmax, swz_xor<8>(vmax));
      const float oldm = mrow[r];
      const float nm   = fmaxf(oldm, vmax);
      mrow[r] = nm;
      const float rs = __expf(oldm - nm);          // 0 on first chunk
      const float p0 = __expf(s0 - nm);
      const float p1 = __expf(s1 - nm);
      const float p2 = __expf(s2 - nm);
      const float p3 = __expf(s3 - nm);
      const int   m  = r + (lane >> 4) * 8;
      pmine[m * LDP +      nn] = (_Float16)p0;
      pmine[m * LDP + 16 + nn] = (_Float16)p1;
      pmine[m * LDP + 32 + nn] = (_Float16)p2;
      pmine[m * LDP + 48 + nn] = (_Float16)p3;
      float rsum = (p0 + p1) + (p2 + p3);
      rsum += swz_xor<1>(rsum);
      rsum += swz_xor<2>(rsum);
      rsum += swz_xor<4>(rsum);
      rsum += swz_xor<8>(rsum);
      lrow[r] = lrow[r] * rs + rsum;
#pragma unroll
      for (int j = 0; j < 4; ++j) acc[j][r] *= rs;
    }

    // P (16x64) · V (64x64): two A fragments, V fragments K-contiguous
    v16h ap0 = load_a_frag(pmine, LDP);        // keys 0..31
    v16h ap1 = load_a_frag(pmine + 32, LDP);   // keys 32..63
#pragma unroll
    for (int j = 0; j < 4; ++j) {
      v16h bv0 = load_b_fragNK(&Vs[cur][(j * 16) * LDV], LDV);
      v16h bv1 = load_b_fragNK(&Vs[cur][(j * 16) * LDV + 32], LDV);
      acc[j] = wmma_f16(ap0, bv0, acc[j]);
      acc[j] = wmma_f16(ap1, bv1, acc[j]);
    }
    __syncthreads();
    cur ^= 1;
  }

  // normalize and store f16 output at (b, q, h*64 + d)
  const int lhalf = lane >> 4;
  float rinv[8];
#pragma unroll
  for (int r = 0; r < 8; ++r) rinv[r] = 1.0f / lrow[r];
#pragma unroll
  for (int j = 0; j < 4; ++j) {
#pragma unroll
    for (int r = 0; r < 8; ++r) {
      const int row = q0 + r + lhalf * 8;
      const float v = acc[j][r] * rinv[r];
      O[((size_t)(b * kS) + row) * kD + h * kDK + j * 16 + nn] = (_Float16)v;
    }
  }
}

// ---------------------------------------------------------------------------
// Residual add + LayerNorm (ddof=1, scalar gamma/beta). One block per row.
// ---------------------------------------------------------------------------
__global__ void __launch_bounds__(256)
enc_ln_kernel(const float* __restrict__ X, const float* __restrict__ Yadd,
              const float* __restrict__ gamma, const float* __restrict__ beta,
              float* __restrict__ out_f32, _Float16* __restrict__ out_f16) {
  const int row  = blockIdx.x;
  const int tid  = threadIdx.x;
  const int wave = tid >> 5;
  const float* x = X    + (size_t)row * kD;
  const float* y = Yadd + (size_t)row * kD;

  __shared__ float red_s[8];
  __shared__ float red_q[8];

  const v4f xv = ((const v4f*)x)[tid];
  const v4f yv = ((const v4f*)y)[tid];
  const v4f v  = xv + yv;
  float sum   = (v.x + v.y) + (v.z + v.w);
  float sumsq = (v.x * v.x + v.y * v.y) + (v.z * v.z + v.w * v.w);
  sum += swz_xor<1>(sum);   sumsq += swz_xor<1>(sumsq);
  sum += swz_xor<2>(sum);   sumsq += swz_xor<2>(sumsq);
  sum += swz_xor<4>(sum);   sumsq += swz_xor<4>(sumsq);
  sum += swz_xor<8>(sum);   sumsq += swz_xor<8>(sumsq);
  sum += swz_xor<16>(sum);  sumsq += swz_xor<16>(sumsq);
  if ((tid & 31) == 0) { red_s[wave] = sum; red_q[wave] = sumsq; }
  __syncthreads();
  float ts = 0.0f, tq = 0.0f;
#pragma unroll
  for (int w = 0; w < 8; ++w) { ts += red_s[w]; tq += red_q[w]; }

  const float mean = ts / (float)kD;
  const float var  = (tq - (float)kD * mean * mean) / (float)(kD - 1);
  const float g    = gamma[0] * rsqrtf(var + kEps);
  const float bt   = beta[0];
  v4f o;
  o.x = (v.x - mean) * g + bt;
  o.y = (v.y - mean) * g + bt;
  o.z = (v.z - mean) * g + bt;
  o.w = (v.w - mean) * g + bt;
  ((v4f*)(out_f32 + (size_t)row * kD))[tid] = o;
  if (out_f16) {
    v4h oh;
    oh[0] = (_Float16)o.x; oh[1] = (_Float16)o.y;
    oh[2] = (_Float16)o.z; oh[3] = (_Float16)o.w;
    ((v4h*)(out_f16 + (size_t)row * kD))[tid] = oh;
  }
}

// ---------------------------------------------------------------------------
// Host: orchestrate the full encoder block on `stream`
// ---------------------------------------------------------------------------
extern "C" void kernel_launch(void* const* d_in, const int* in_sizes, int n_in,
                              void* d_out, int out_size, void* d_ws, size_t ws_size,
                              hipStream_t stream) {
  (void)in_sizes; (void)n_in; (void)out_size; (void)ws_size;

  const float* x   = (const float*)d_in[0];
  /* d_in[1] = src_mask: provably a no-op in the reference */
  const float* wq  = (const float*)d_in[2];
  const float* wk  = (const float*)d_in[3];
  const float* wv  = (const float*)d_in[4];
  const float* wo  = (const float*)d_in[5];
  const float* w1  = (const float*)d_in[6];
  const float* b1  = (const float*)d_in[7];
  const float* w2  = (const float*)d_in[8];
  const float* b2  = (const float*)d_in[9];
  const float* g1  = (const float*)d_in[10];
  const float* be1 = (const float*)d_in[11];
  const float* g2  = (const float*)d_in[12];
  const float* be2 = (const float*)d_in[13];

  char* ws = (char*)d_ws;
  size_t off = 0;
  auto alloc = [&](size_t bytes) -> char* {
    char* p = ws + off;
    off += (bytes + 255) & ~(size_t)255;
    return p;
  };

  _Float16* xh   = (_Float16*)alloc((size_t)kM * kD * 2);
  _Float16* wqt  = (_Float16*)alloc((size_t)kD * kD * 2);
  _Float16* wkt  = (_Float16*)alloc((size_t)kD * kD * 2);
  _Float16* wvt  = (_Float16*)alloc((size_t)kD * kD * 2);
  _Float16* wot  = (_Float16*)alloc((size_t)kD * kD * 2);
  _Float16* w1t  = (_Float16*)alloc((size_t)kD * kDFF * 2);   // [DFF][D]
  _Float16* w2t  = (_Float16*)alloc((size_t)kDFF * kD * 2);   // [D][DFF]
  _Float16* qh   = (_Float16*)alloc((size_t)kM * kD * 2);     // pre-scaled Q
  _Float16* kh   = (_Float16*)alloc((size_t)kM * kD * 2);
  _Float16* vt   = (_Float16*)alloc((size_t)kM * kD * 2);     // [(b,h,d)][s]
  _Float16* oh   = (_Float16*)alloc((size_t)kM * kD * 2);
  float*    proj = (float*)   alloc((size_t)kM * kD * 4);
  float*    x1   = (float*)   alloc((size_t)kM * kD * 4);
  _Float16* x1h  = (_Float16*)alloc((size_t)kM * kD * 2);
  // Aliases (lifetimes verified): FFN1 output (4096x4096 f16 = 32MB) reuses
  // the q/k/v/o region (fully consumed by then); FFN2 output reuses proj.
  _Float16* ff1h   = qh;
  float*    ff2out = proj;

  const dim3 blk(256);

  // 1) stage activations (f16) and weights (f16, transposed to [N][K])
  enc_cvt_f16_kernel<<<dim3((kM * kD + 255) / 256), blk, 0, stream>>>(
      x, xh, kM * kD);
  enc_transpose_cvt_kernel<<<dim3(kD / 32, kD / 32), blk, 0, stream>>>(wq, wqt, kD, kD);
  enc_transpose_cvt_kernel<<<dim3(kD / 32, kD / 32), blk, 0, stream>>>(wk, wkt, kD, kD);
  enc_transpose_cvt_kernel<<<dim3(kD / 32, kD / 32), blk, 0, stream>>>(wv, wvt, kD, kD);
  enc_transpose_cvt_kernel<<<dim3(kD / 32, kD / 32), blk, 0, stream>>>(wo, wot, kD, kD);
  enc_transpose_cvt_kernel<<<dim3(kDFF / 32, kD / 32), blk, 0, stream>>>(w1, w1t, kD, kDFF);
  enc_transpose_cvt_kernel<<<dim3(kD / 32, kDFF / 32), blk, 0, stream>>>(w2, w2t, kDFF, kD);

  // 2) Q (pre-scaled by 1/8) / K projections (f16), V projection (transposed)
  {
    dim3 grid(kD / 128, kM / 128);
    enc_gemm_kernel<0, 0, 1, 1><<<grid, blk, 0, stream>>>(xh, wqt, nullptr, qh, kM, kD, kD);
    enc_gemm_kernel<0, 0, 1, 0><<<grid, blk, 0, stream>>>(xh, wkt, nullptr, kh, kM, kD, kD);
    enc_gemm_kernel<0, 0, 2, 0><<<grid, blk, 0, stream>>>(xh, wvt, nullptr, vt, kM, kD, kD);
  }
  // 3) attention
  enc_attention_kernel<<<dim3(kS / 128, kH, kB), blk, 0, stream>>>(qh, kh, vt, oh);
  // 4) output projection (f32)
  enc_gemm_kernel<0, 0, 0, 0><<<dim3(kD / 128, kM / 128), blk, 0, stream>>>(
      oh, wot, nullptr, proj, kM, kD, kD);
  // 5) residual + LN1 -> x1 (f32) and x1h (f16)
  enc_ln_kernel<<<dim3(kM), blk, 0, stream>>>(x, proj, g1, be1, x1, x1h);
  // 6) FFN1: relu(x1 @ w1 + b1) -> f16
  enc_gemm_kernel<1, 1, 1, 0><<<dim3(kDFF / 128, kM / 128), blk, 0, stream>>>(
      x1h, w1t, b1, ff1h, kM, kDFF, kD);
  // 7) FFN2: ff1 @ w2 + b2 -> f32
  enc_gemm_kernel<1, 0, 0, 0><<<dim3(kD / 128, kM / 128), blk, 0, stream>>>(
      ff1h, w2t, b2, ff2out, kM, kD, kDFF);
  // 8) residual + LN2 -> d_out (f32)
  enc_ln_kernel<<<dim3(kM), blk, 0, stream>>>(x1, ff2out, g2, be2, (float*)d_out, nullptr);
}
